// SimpleMamba_20641612825354
// MI455X (gfx1250) — matrix operand order repaired and checked by reading, not compiled, not run on hardware
//
#include <hip/hip_runtime.h>
#include <hip/hip_bf16.h>
#include <stdint.h>
#include <stddef.h>

// Problem sizes (fixed by reference)
#define B_    4
#define L_    2048
#define DIN_  128
#define H_    512
#define NCLS_ 10
#define DI_   1024          // D_INNER
#define NS_   16            // D_STATE
#define DC_   4             // D_CONV
#define DTR_  32            // DT_RANK
#define TOK_  (B_ * L_)     // 8192
#define DBCW_ (DTR_ + 2 * NS_)  // 64

typedef unsigned short bf16_t;
typedef __attribute__((ext_vector_type(16))) __bf16        v16bf;
typedef __attribute__((ext_vector_type(8)))  float         v8f;
typedef __attribute__((ext_vector_type(4)))  unsigned int  u32x4;  // trivial 16B vector

// ---- force global (AS1) memory ops via inttoptr (kills flat_load/flat_store) ----
typedef const __attribute__((address_space(1))) u32x4*   gcu4p;
typedef __attribute__((address_space(1))) bf16_t*        gbf16p;
typedef __attribute__((address_space(1))) float*         gf32p;

__device__ __forceinline__ u32x4 gload_u4(const void* p) {
    return *(gcu4p)(unsigned long long)p;
}
__device__ __forceinline__ void gstore_bf16(bf16_t* p, bf16_t v) {
    *(gbf16p)(unsigned long long)p = v;
}
__device__ __forceinline__ void gstore_f32(float* p, float v) {
    *(gf32p)(unsigned long long)p = v;
}

__device__ __forceinline__ float bf2f(bf16_t v) {
    unsigned u = ((unsigned)v) << 16;
    return __builtin_bit_cast(float, u);
}
__device__ __forceinline__ bf16_t f2bf(float f) {
    unsigned u = __builtin_bit_cast(unsigned, f);
    unsigned r = u + 0x7FFFu + ((u >> 16) & 1u);   // round-to-nearest-even
    return (bf16_t)(r >> 16);
}

// ---------------- elementwise prep ----------------
__global__ void cvt_f32_bf16_kernel(const float* __restrict__ src,
                                    bf16_t* __restrict__ dst, int n) {
    int i = blockIdx.x * blockDim.x + threadIdx.x;
    if (i < n) dst[i] = f2bf(src[i]);
}

__global__ void negexp_kernel(const float* __restrict__ src,
                              float* __restrict__ dst, int n) {
    int i = blockIdx.x * blockDim.x + threadIdx.x;
    if (i < n) dst[i] = -__expf(src[i]);
}

// ---------------- WMMA bf16 GEMM: C[M,N] = A[M,K] @ W[N,K]^T ----------------
// One wave computes a 16(M) x 64(N) tile: 1 A-fragment x 4 B-fragments,
// 4x v_wmma_f32_16x16x32_bf16 per K-step of 32.  Software-pipelined: the
// loads for k+32 are issued before the WMMAs for k, so the matrix pipe never
// waits on the loads issued in the same iteration.
union Frag { u32x4 u[2]; v16bf v; };

template <bool OUT_BF16, bool BIAS, bool SOFTPLUS>
__global__ void wmma_gemm_kernel(const bf16_t* __restrict__ A,
                                 const bf16_t* __restrict__ W,
                                 const float* __restrict__ bias,
                                 void* __restrict__ Cout,
                                 int M, int N, int K) {
    const int gtid  = blockIdx.x * blockDim.x + threadIdx.x;
    const int wave  = gtid >> 5;
    const int lane  = gtid & 31;
    const int strips = N >> 6;                 // 64-wide N strips
    const int m0 = (wave / strips) << 4;
    const int n0 = (wave % strips) << 6;
    if (m0 >= M) return;

    const int half = lane >> 4;                // lane-half: K split per ISA layout
    const int lr   = lane & 15;

    v8f acc[4];
    acc[0] = {}; acc[1] = {}; acc[2] = {}; acc[3] = {};

    // A: lane (lr) = row m; half selects K {0-7,16-23} vs {8-15,24-31}
    const size_t aoff = (size_t)(m0 + lr) * K + half * 8;
    // B = W^T: column n of B = row n of W (K-contiguous); half selects K {0-7,16-23} vs {8-15,24-31}
    size_t boff[4];
#pragma unroll
    for (int j = 0; j < 4; ++j)
        boff[j] = (size_t)(n0 + j * 16 + lr) * K + half * 8;

    Frag a_cur, b_cur[4];
    // prologue: load k-block 0
    a_cur.u[0] = gload_u4(A + aoff);
    a_cur.u[1] = gload_u4(A + aoff + 16);
#pragma unroll
    for (int j = 0; j < 4; ++j) {
        b_cur[j].u[0] = gload_u4(W + boff[j]);
        b_cur[j].u[1] = gload_u4(W + boff[j] + 16);
    }

    int k0 = 0;
    for (; k0 + 32 < K; k0 += 32) {
        Frag a_nxt, b_nxt[4];
        const int kn = k0 + 32;
        a_nxt.u[0] = gload_u4(A + aoff + kn);
        a_nxt.u[1] = gload_u4(A + aoff + kn + 16);
#pragma unroll
        for (int j = 0; j < 4; ++j) {
            b_nxt[j].u[0] = gload_u4(W + boff[j] + kn);
            b_nxt[j].u[1] = gload_u4(W + boff[j] + kn + 16);
        }
        __builtin_prefetch(A + aoff + kn + 64, 0, 3);
#pragma unroll
        for (int j = 0; j < 4; ++j) {
            acc[j] = __builtin_amdgcn_wmma_f32_16x16x32_bf16(
                false, a_cur.v, false, b_cur[j].v, (short)0, acc[j], false, false);
        }
        a_cur = a_nxt;
#pragma unroll
        for (int j = 0; j < 4; ++j) b_cur[j] = b_nxt[j];
    }
    // epilogue k-block
#pragma unroll
    for (int j = 0; j < 4; ++j) {
        acc[j] = __builtin_amdgcn_wmma_f32_16x16x32_bf16(
            false, a_cur.v, false, b_cur[j].v, (short)0, acc[j], false, false);
    }

    // D layout: lane n = lr, rows m = half*8 + v (v = accumulator element)
    const int mrow = m0 + half * 8;
#pragma unroll
    for (int j = 0; j < 4; ++j) {
        const int n = n0 + j * 16 + lr;
        const float bv = BIAS ? bias[n] : 0.0f;
#pragma unroll
        for (int v = 0; v < 8; ++v) {
            float c = acc[j][v] + bv;
            if (SOFTPLUS) c = (c > 20.0f) ? c : log1pf(__expf(c));
            const size_t idx = (size_t)(mrow + v) * N + n;
            if (OUT_BF16) gstore_bf16((bf16_t*)Cout + idx, f2bf(c));
            else          gstore_f32((float*)Cout + idx, c);
        }
    }
}

// ---------------- causal depthwise conv1d + SiLU ----------------
__global__ void conv_silu_kernel(const bf16_t* __restrict__ xz,   // [TOK, 2*DI]
                                 const float* __restrict__ conv_w, // [DI, DC]
                                 const float* __restrict__ conv_b, // [DI]
                                 bf16_t* __restrict__ xc) {        // [TOK, DI]
    int i = blockIdx.x * blockDim.x + threadIdx.x;
    if (i >= TOK_ * DI_) return;
    int d   = i & (DI_ - 1);
    int tok = i >> 10;
    int t   = tok & (L_ - 1);
    float s = conv_b[d];
#pragma unroll
    for (int k = 0; k < DC_; ++k) {
        int tt = t - (DC_ - 1) + k;
        if (tt >= 0)
            s += conv_w[d * DC_ + k] *
                 bf2f(xz[(size_t)(tok - (DC_ - 1) + k) * (2 * DI_) + d]);
    }
    float sig = 1.0f / (1.0f + __expf(-s));
    gstore_bf16(xc + (size_t)tok * DI_ + d, f2bf(s * sig));
}

// ---------------- extract dt columns of dbc as compact bf16 ----------------
__global__ void extract_dt_kernel(const float* __restrict__ dbc,  // [TOK, 64]
                                  bf16_t* __restrict__ dt_bf) {   // [TOK, 32]
    int i = blockIdx.x * blockDim.x + threadIdx.x;
    if (i >= TOK_ * DTR_) return;
    int c   = i & (DTR_ - 1);
    int tok = i >> 5;
    dt_bf[i] = f2bf(dbc[(size_t)tok * DBCW_ + c]);
}

// ---------------- selective scan (only last-step y is needed) ----------------
__global__ void scan_kernel(const bf16_t* __restrict__ delta,  // [TOK, DI]
                            const bf16_t* __restrict__ xc,     // [TOK, DI]
                            const float* __restrict__ dbc,     // [TOK, 64]
                            const float* __restrict__ Aneg,    // [DI, NS]
                            const bf16_t* __restrict__ xz,     // [TOK, 2*DI]
                            const float* __restrict__ Dvec,    // [DI]
                            float* __restrict__ y_last) {      // [B, DI]
    int tid = blockIdx.x * blockDim.x + threadIdx.x;  // 65536 threads
    int n = tid & (NS_ - 1);
    int d = (tid >> 4) & (DI_ - 1);
    int b = tid >> 14;
    const float a = Aneg[d * NS_ + n];
    const bf16_t* dp = delta + (size_t)b * L_ * DI_ + d;
    const bf16_t* xp = xc    + (size_t)b * L_ * DI_ + d;
    const float*  bp = dbc   + (size_t)b * L_ * DBCW_ + DTR_ + n;
    float h = 0.0f;
    for (int t = 0; t < L_; ++t) {
        float dl = bf2f(dp[(size_t)t * DI_]);
        float xv = bf2f(xp[(size_t)t * DI_]);
        float bv = bp[(size_t)t * DBCW_];
        h = __expf(dl * a) * h + (dl * xv) * bv;
    }
    const size_t lastTok = (size_t)b * L_ + (L_ - 1);
    float c = dbc[lastTok * DBCW_ + DTR_ + NS_ + n];
    float p = h * c;
    p += __shfl_xor(p, 1, 32);
    p += __shfl_xor(p, 2, 32);
    p += __shfl_xor(p, 4, 32);
    p += __shfl_xor(p, 8, 32);
    if (n == 0) {
        float xcl = bf2f(xc[lastTok * DI_ + d]);
        float zl  = bf2f(xz[lastTok * (2 * DI_) + DI_ + d]);
        float y = (p + xcl * Dvec[d]) * (zl / (1.0f + __expf(-zl)));
        gstore_f32(y_last + b * DI_ + d, y);
    }
}

// ---------------- head: out_proj at last step, then fc ----------------
__global__ void out_proj_kernel(const float* __restrict__ y_last,   // [B, DI]
                                const float* __restrict__ opw,      // [H, DI]
                                float* __restrict__ out_last) {     // [B, H]
    int i = blockIdx.x * blockDim.x + threadIdx.x;
    if (i >= B_ * H_) return;
    int hc = i % H_, b = i / H_;
    const float* y = y_last + b * DI_;
    const float* w = opw + (size_t)hc * DI_;
    float s = 0.0f;
    for (int k = 0; k < DI_; ++k) s += y[k] * w[k];
    gstore_f32(out_last + i, s);
}

__global__ void fc_kernel(const float* __restrict__ out_last, // [B, H]
                          const float* __restrict__ fcw,      // [NCLS, H]
                          const float* __restrict__ fcb,      // [NCLS]
                          float* __restrict__ out) {          // [B, NCLS]
    int i = blockIdx.x * blockDim.x + threadIdx.x;
    if (i >= B_ * NCLS_) return;
    int c = i % NCLS_, b = i / NCLS_;
    float s = fcb[c];
    for (int k = 0; k < H_; ++k) s += out_last[b * H_ + k] * fcw[(size_t)c * H_ + k];
    gstore_f32(out + i, s);
}

// ---------------- launcher ----------------
static inline size_t alignup(size_t v) { return (v + 255) & ~(size_t)255; }

extern "C" void kernel_launch(void* const* d_in, const int* in_sizes, int n_in,
                              void* d_out, int out_size, void* d_ws, size_t ws_size,
                              hipStream_t stream) {
    const float* x         = (const float*)d_in[0];   // [B,L,DIN]
    const float* emb_w     = (const float*)d_in[1];   // [H,DIN]
    const float* emb_b     = (const float*)d_in[2];   // [H]
    const float* in_proj_w = (const float*)d_in[3];   // [2*DI,H]
    const float* conv_w    = (const float*)d_in[4];   // [DI,DC]
    const float* conv_b    = (const float*)d_in[5];   // [DI]
    const float* x_proj_w  = (const float*)d_in[6];   // [64,DI]
    const float* dt_proj_w = (const float*)d_in[7];   // [DI,DTR]
    const float* dt_proj_b = (const float*)d_in[8];   // [DI]
    const float* A_log     = (const float*)d_in[9];   // [DI,NS]
    const float* Dvec      = (const float*)d_in[10];  // [DI]
    const float* out_prj_w = (const float*)d_in[11];  // [H,DI]
    const float* fc_w      = (const float*)d_in[12];  // [NCLS,H]
    const float* fc_b      = (const float*)d_in[13];  // [NCLS]
    float* outp = (float*)d_out;                      // [B,NCLS] f32

    // workspace layout (~82 MB)
    char* ws = (char*)d_ws;
    size_t off = 0;
    bf16_t* x_bf   = (bf16_t*)(ws + off); off = alignup(off + (size_t)TOK_ * DIN_ * 2);
    bf16_t* embw_bf= (bf16_t*)(ws + off); off = alignup(off + (size_t)H_ * DIN_ * 2);
    bf16_t* ipw_bf = (bf16_t*)(ws + off); off = alignup(off + (size_t)2 * DI_ * H_ * 2);
    bf16_t* xpw_bf = (bf16_t*)(ws + off); off = alignup(off + (size_t)DBCW_ * DI_ * 2);
    bf16_t* dtw_bf = (bf16_t*)(ws + off); off = alignup(off + (size_t)DI_ * DTR_ * 2);
    float*  Aneg   = (float*) (ws + off); off = alignup(off + (size_t)DI_ * NS_ * 4);
    bf16_t* h_bf   = (bf16_t*)(ws + off); off = alignup(off + (size_t)TOK_ * H_ * 2);
    bf16_t* xz_bf  = (bf16_t*)(ws + off); off = alignup(off + (size_t)TOK_ * 2 * DI_ * 2);
    bf16_t* xc_bf  = (bf16_t*)(ws + off); off = alignup(off + (size_t)TOK_ * DI_ * 2);
    float*  dbc    = (float*) (ws + off); off = alignup(off + (size_t)TOK_ * DBCW_ * 4);
    bf16_t* dt_bf  = (bf16_t*)(ws + off); off = alignup(off + (size_t)TOK_ * DTR_ * 2);
    bf16_t* del_bf = (bf16_t*)(ws + off); off = alignup(off + (size_t)TOK_ * DI_ * 2);
    float*  y_last = (float*) (ws + off); off = alignup(off + (size_t)B_ * DI_ * 4);
    float*  o_last = (float*) (ws + off); off = alignup(off + (size_t)B_ * H_ * 4);
    (void)ws_size; (void)n_in; (void)in_sizes; (void)out_size;

    const int TB = 256;
    auto blk = [](int n, int tb) { return (n + tb - 1) / tb; };

    // precision conversion + A = -exp(A_log)
    cvt_f32_bf16_kernel<<<blk(TOK_*DIN_,TB), TB, 0, stream>>>(x, x_bf, TOK_*DIN_);
    cvt_f32_bf16_kernel<<<blk(H_*DIN_,TB),   TB, 0, stream>>>(emb_w, embw_bf, H_*DIN_);
    cvt_f32_bf16_kernel<<<blk(2*DI_*H_,TB),  TB, 0, stream>>>(in_proj_w, ipw_bf, 2*DI_*H_);
    cvt_f32_bf16_kernel<<<blk(DBCW_*DI_,TB), TB, 0, stream>>>(x_proj_w, xpw_bf, DBCW_*DI_);
    cvt_f32_bf16_kernel<<<blk(DI_*DTR_,TB),  TB, 0, stream>>>(dt_proj_w, dtw_bf, DI_*DTR_);
    negexp_kernel<<<blk(DI_*NS_,TB),         TB, 0, stream>>>(A_log, Aneg, DI_*NS_);

    // G1: h = x @ emb_w^T + emb_b   [8192,512] (bf16 out)
    {
        int waves = (TOK_/16) * (H_/64);
        wmma_gemm_kernel<true, true, false><<<waves*32/TB, TB, 0, stream>>>(
            x_bf, embw_bf, emb_b, h_bf, TOK_, H_, DIN_);
    }
    // G2: xz = h @ in_proj_w^T      [8192,2048] (bf16 out)
    {
        int waves = (TOK_/16) * ((2*DI_)/64);
        wmma_gemm_kernel<true, false, false><<<waves*32/TB, TB, 0, stream>>>(
            h_bf, ipw_bf, nullptr, xz_bf, TOK_, 2*DI_, H_);
    }
    // conv + SiLU -> xc (bf16)
    conv_silu_kernel<<<blk(TOK_*DI_,TB), TB, 0, stream>>>(xz_bf, conv_w, conv_b, xc_bf);
    // G4: dbc = xc @ x_proj_w^T     [8192,64] (f32 out)
    {
        int waves = (TOK_/16) * (DBCW_/64);
        wmma_gemm_kernel<false, false, false><<<waves*32/TB, TB, 0, stream>>>(
            xc_bf, xpw_bf, nullptr, dbc, TOK_, DBCW_, DI_);
    }
    // compact dt columns to bf16
    extract_dt_kernel<<<blk(TOK_*DTR_,TB), TB, 0, stream>>>(dbc, dt_bf);
    // G5: delta = softplus(dt @ dt_proj_w^T + b)  [8192,1024] (bf16 out)
    {
        int waves = (TOK_/16) * (DI_/64);
        wmma_gemm_kernel<true, true, true><<<waves*32/TB, TB, 0, stream>>>(
            dt_bf, dtw_bf, dt_proj_b, del_bf, TOK_, DI_, DTR_);
    }
    // selective scan (L2-resident recurrence), fused skip + gate at t = L-1
    scan_kernel<<<(B_*DI_*NS_)/TB, TB, 0, stream>>>(
        del_bf, xc_bf, dbc, Aneg, xz_bf, Dvec, y_last);
    // head
    out_proj_kernel<<<blk(B_*H_,TB), TB, 0, stream>>>(y_last, out_prj_w, o_last);
    fc_kernel<<<1, 64, 0, stream>>>(o_last, fc_w, fc_b, outp);
}
